// DrpModel_88304527606469
// MI455X (gfx1250) — matrix operand order repaired and checked by reading, compiled, not run on hardware
//
#include <hip/hip_runtime.h>

#define N_NODES 400000
#define N_EDGES 1600000
#define NG      8192
#define DD      128
#define GENES   954
#define GENES_P 960
#define BN_EPS  1e-5f
#define TPB     256

typedef __attribute__((ext_vector_type(16))) __bf16 v16bf;
typedef __attribute__((ext_vector_type(8)))  float  v8f;

union FragU { v16bf v; uint4 q[2]; };

__device__ __forceinline__ unsigned short f2bf(float f) {
  unsigned int u = __float_as_uint(f);
  u += 0x7fffu + ((u >> 16) & 1u);           // round-to-nearest-even
  return (unsigned short)(u >> 16);
}

// ---------------------------------------------------------------------------
// Generic bf16 WMMA GEMM: out = act(A[M,K] @ W[K,Nc] + bias)
// A: bf16 row-major [M][K].  Wt: bf16 row-major [Nc][K] (i.e. W transposed).
// Block = 8 waves; tile = 16 rows x 128 cols; A slab staged in LDS via
// CDNA5 async global->LDS b128 transfers (ASYNCcnt), bypassing VGPRs.
// ---------------------------------------------------------------------------
template<bool RELU, bool OBF>
__global__ __launch_bounds__(TPB) void gemm_bf16_kernel(
    const unsigned short* __restrict__ A,
    const unsigned short* __restrict__ Wt,
    const float* __restrict__ bias,
    void* __restrict__ outp,
    int M, int Nc, int K, int ldc)
{
  __shared__ __align__(16) unsigned short Asl[16 * 1024];   // up to K=1024 (32 KB)
  const int tid  = threadIdx.x;
  const int lane = tid & 31;
  const int wave = tid >> 5;
  const int rowBase = blockIdx.x * 16;
  const int colTile = blockIdx.y * 8 + wave;  // 16-wide column tile
  const int colBase = colTile * 16;

  // Async cooperative load of the 16 x K A-slab (contiguous in memory):
  // global_load_async_to_lds_b128 writes LDS directly (no VGPR round-trip).
  {
    const char* src = (const char*)(A + (size_t)rowBase * K);
    const unsigned ldsBase = (unsigned)(uintptr_t)(&Asl[0]); // flat[31:0] = LDS offset
    const int total = (16 * K * 2) >> 4;      // number of b128 transfers
    for (int i = tid; i < total; i += TPB) {
      unsigned long long ga = (unsigned long long)(src + ((size_t)i << 4));
      unsigned la = ldsBase + (i << 4);
      asm volatile("global_load_async_to_lds_b128 %0, %1, off"
                   :: "v"(la), "v"(ga) : "memory");
    }
    asm volatile("s_wait_asynccnt 0" ::: "memory");  // barrier does not cover ASYNCcnt
  }
  __syncthreads();

  const int r    = lane & 15;
  const int half = lane >> 4;
  const unsigned short* arow = Asl + r * K;
  const unsigned short* brow = Wt + (size_t)(colBase + r) * K;
  __builtin_prefetch(brow, 0, 0);             // global_prefetch of B row

  v8f acc = {};
  for (int k0 = 0; k0 < K; k0 += 32) {
    FragU a, b;
    // A frag (ISA 16-bit A 16x32): lanes 0-15 hold K 0-7 & 16-23, lanes 16-31 K 8-15 & 24-31
    const uint4* as = (const uint4*)(arow + k0 + (half << 3));
    a.q[0] = as[0];
    a.q[1] = as[2];
    // B frag (K x 16, lane = column): lanes 0-15 hold K 0-7 & 16-23, lanes 16-31 K 8-15 & 24-31
    const uint4* bs = (const uint4*)(brow + k0 + (half << 3));
    b.q[0] = bs[0];
    b.q[1] = bs[2];
    acc = __builtin_amdgcn_wmma_f32_16x16x32_bf16(false, a.v, false, b.v,
                                                  (short)0, acc, false, false);
  }

  const int n  = colBase + r;
  const float bv = bias[n];
#pragma unroll
  for (int i = 0; i < 8; ++i) {               // C layout: vgpr i -> row i + 8*half
    const int m = rowBase + i + (half << 3);
    float v = acc[i] + bv;
    if (RELU) v = v > 0.f ? v : 0.f;
    if (OBF) ((unsigned short*)outp)[(size_t)m * ldc + n] = f2bf(v);
    else     ((float*)outp)[(size_t)m * ldc + n] = v;
  }
}

// ---------------------------------------------------------------------------
// Elementwise / data-movement kernels
// ---------------------------------------------------------------------------
__global__ void embed_kernel(const int* __restrict__ x, const float* __restrict__ emb,
                             float* __restrict__ h) {
  int t = blockIdx.x * TPB + threadIdx.x;     // N*32 threads, 4 floats each
  int n = t >> 5, q = t & 31;
  if (n >= N_NODES) return;
  float4 v = ((const float4*)(emb + (size_t)x[n] * DD))[q];
  ((float4*)(h + (size_t)n * DD))[q] = v;
}

__global__ void scatter_kernel(const int* __restrict__ ei, const float* __restrict__ h,
                               float* __restrict__ agg) {
  int t = blockIdx.x * TPB + threadIdx.x;     // E*32 threads
  int e = t >> 5, q = t & 31;
  if (e >= N_EDGES) return;
  int s = ei[e], d = ei[N_EDGES + e];
  float4 v = ((const float4*)(h + (size_t)s * DD))[q];
  float* p = agg + (size_t)d * DD + q * 4;
  atomicAdd(p + 0, v.x); atomicAdd(p + 1, v.y);
  atomicAdd(p + 2, v.z); atomicAdd(p + 3, v.w);
}

__global__ void pool_kernel(const int* __restrict__ batch, const float* __restrict__ h,
                            float* __restrict__ pooled) {
  int t = blockIdx.x * TPB + threadIdx.x;
  int n = t >> 5, q = t & 31;
  if (n >= N_NODES) return;
  int g = batch[n];
  float4 v = ((const float4*)(h + (size_t)n * DD))[q];
  float* p = pooled + (size_t)g * DD + q * 4;
  atomicAdd(p + 0, v.x); atomicAdd(p + 1, v.y);
  atomicAdd(p + 2, v.z); atomicAdd(p + 3, v.w);
}

// f32 [rows][K] -> bf16 [rows][Kpad] (zero pad), also plain convert when K==Kpad
__global__ void cvt_bf16_kernel(const float* __restrict__ in, unsigned short* __restrict__ out,
                                int rows, int K, int Kpad) {
  int idx = blockIdx.x * TPB + threadIdx.x;
  if (idx >= rows * Kpad) return;
  int r = idx / Kpad, k = idx - r * Kpad;
  out[idx] = (k < K) ? f2bf(in[(size_t)r * K + k]) : (unsigned short)0;
}

// W [K][Nc] f32 -> Wt [Nc][Kpad] bf16 (transpose + pad)
__global__ void cvt_wt_kernel(const float* __restrict__ W, unsigned short* __restrict__ Wt,
                              int K, int Nc, int Kpad) {
  int idx = blockIdx.x * TPB + threadIdx.x;
  if (idx >= Nc * Kpad) return;
  int n = idx / Kpad, k = idx - n * Kpad;
  Wt[idx] = (k < K) ? f2bf(W[(size_t)k * Nc + n]) : (unsigned short)0;
}

__global__ void zero_kernel(float* __restrict__ p, int n) {
  int i = blockIdx.x * TPB + threadIdx.x;
  if (i < n) p[i] = 0.f;
}

__global__ void bn_stats_kernel(const float* __restrict__ h, float* __restrict__ stats,
                                int M, int rpb) {
  int d = threadIdx.x;                        // 128 threads = columns
  int r0 = blockIdx.x * rpb;
  int r1 = r0 + rpb; if (r1 > M) r1 = M;
  float s = 0.f, q = 0.f;
  for (int r = r0; r < r1; ++r) { float v = h[(size_t)r * DD + d]; s += v; q += v * v; }
  atomicAdd(&stats[d], s);
  atomicAdd(&stats[DD + d], q);
}

__global__ void bn_apply_kernel(const float* __restrict__ in, const float* __restrict__ stats,
                                const float* __restrict__ g, const float* __restrict__ b,
                                float* __restrict__ out, int M) {
  int idx = blockIdx.x * TPB + threadIdx.x;
  if (idx >= M * DD) return;
  int d = idx & (DD - 1);
  float inv = 1.f / (float)M;
  float mu  = stats[d] * inv;
  float var = stats[DD + d] * inv - mu * mu;
  float sc  = rsqrtf(var + BN_EPS) * g[d];
  out[idx] = (in[idx] - mu) * sc + b[d];
}

__global__ void gelu_bf16_kernel(const float* __restrict__ in, unsigned short* __restrict__ out,
                                 int n) {
  int idx = blockIdx.x * TPB + threadIdx.x;
  if (idx >= n) return;
  float x = in[idx];
  float c = 0.79788456080286535588f * (x + 0.044715f * x * x * x);
  out[idx] = f2bf(0.5f * x * (1.f + tanhf(c)));
}

__global__ void fc3_kernel(const float* __restrict__ in, const float* __restrict__ w,
                           const float* __restrict__ b, float* __restrict__ out) {
  int row  = blockIdx.x * 8 + (threadIdx.x >> 5);
  int lane = threadIdx.x & 31;
  const float* r = in + (size_t)row * 1024;
  float s = 0.f;
  for (int k = lane; k < 1024; k += 32) s += r[k] * w[k];
  for (int off = 16; off > 0; off >>= 1) s += __shfl_xor(s, off, 32);
  if (lane == 0) out[row] = s + b[0];
}

// ---------------------------------------------------------------------------
static inline void run_gemm(const unsigned short* A, const unsigned short* Wt,
                            const float* bias, void* out, int M, int Nc, int K, int ldc,
                            bool relu, bool obf, hipStream_t s) {
  dim3 grid(M / 16, Nc / 128);
  if (relu && obf)
    gemm_bf16_kernel<true, true ><<<grid, TPB, 0, s>>>(A, Wt, bias, out, M, Nc, K, ldc);
  else if (relu)
    gemm_bf16_kernel<true, false><<<grid, TPB, 0, s>>>(A, Wt, bias, out, M, Nc, K, ldc);
  else if (obf)
    gemm_bf16_kernel<false, true><<<grid, TPB, 0, s>>>(A, Wt, bias, out, M, Nc, K, ldc);
  else
    gemm_bf16_kernel<false, false><<<grid, TPB, 0, s>>>(A, Wt, bias, out, M, Nc, K, ldc);
}

extern "C" void kernel_launch(void* const* d_in, const int* in_sizes, int n_in,
                              void* d_out, int out_size, void* d_ws, size_t ws_size,
                              hipStream_t stream) {
  (void)in_sizes; (void)n_in; (void)out_size; (void)ws_size;

  const float* exp_in = (const float*)d_in[0];
  const int*   x      = (const int*)d_in[1];
  const int*   ei     = (const int*)d_in[2];
  const int*   batch  = (const int*)d_in[3];
  const float* emb    = (const float*)d_in[4];
  const float *w1[5], *b1[5], *w2[5], *b2[5], *bng[5], *bnb[5];
  int p = 5;
  for (int i = 0; i < 5; ++i) {
    w1[i] = (const float*)d_in[p++]; b1[i] = (const float*)d_in[p++];
    w2[i] = (const float*)d_in[p++]; b2[i] = (const float*)d_in[p++];
  }
  for (int i = 0; i < 5; ++i) {
    bng[i] = (const float*)d_in[p++]; bnb[i] = (const float*)d_in[p++];
  }
  const float* fxd_w = (const float*)d_in[35]; const float* fxd_b = (const float*)d_in[36];
  const float* cw1   = (const float*)d_in[37]; const float* cb1   = (const float*)d_in[38];
  const float* cw2   = (const float*)d_in[39]; const float* cb2   = (const float*)d_in[40];
  const float* fc1w  = (const float*)d_in[41]; const float* fc1b  = (const float*)d_in[42];
  const float* fc2w  = (const float*)d_in[43]; const float* fc2b  = (const float*)d_in[44];
  const float* fc3w  = (const float*)d_in[45]; const float* fc3b  = (const float*)d_in[46];

  // -------- workspace layout --------
  const size_t ND = (size_t)N_NODES * DD;
  char* wp = (char*)d_ws;
  float* bufA = (float*)wp;                 wp += ND * sizeof(float);
  float* bufB = (float*)wp;                 wp += ND * sizeof(float);
  unsigned short* bufBF = (unsigned short*)wp; wp += ND * sizeof(unsigned short);
  auto allocU = [&](size_t n) { unsigned short* r = (unsigned short*)wp; wp += n * 2; return r; };
  unsigned short *w1t[5], *w2t[5];
  for (int i = 0; i < 5; ++i) { w1t[i] = allocU(DD * DD); w2t[i] = allocU(DD * DD); }
  unsigned short* fxdt  = allocU(DD * DD);
  unsigned short* cw1t  = allocU((size_t)DD * GENES_P);
  unsigned short* cw2t  = allocU(DD * DD);
  unsigned short* fc1t  = allocU((size_t)1024 * 256);
  unsigned short* fc2t  = allocU((size_t)1024 * 1024);
  unsigned short* expbf = allocU((size_t)NG * GENES_P);
  float* stats  = (float*)wp;               wp += 256 * sizeof(float);
  float* pooled = (float*)wp;               wp += (size_t)NG * DD * sizeof(float);

  // -------- weight / input conversion to bf16 (transposed, K-padded) --------
  for (int i = 0; i < 5; ++i) {
    cvt_wt_kernel<<<(DD * DD + TPB - 1) / TPB, TPB, 0, stream>>>(w1[i], w1t[i], DD, DD, DD);
    cvt_wt_kernel<<<(DD * DD + TPB - 1) / TPB, TPB, 0, stream>>>(w2[i], w2t[i], DD, DD, DD);
  }
  cvt_wt_kernel<<<(DD * DD + TPB - 1) / TPB, TPB, 0, stream>>>(fxd_w, fxdt, DD, DD, DD);
  cvt_wt_kernel<<<(DD * GENES_P + TPB - 1) / TPB, TPB, 0, stream>>>(cw1, cw1t, GENES, DD, GENES_P);
  cvt_wt_kernel<<<(DD * DD + TPB - 1) / TPB, TPB, 0, stream>>>(cw2, cw2t, DD, DD, DD);
  cvt_wt_kernel<<<(1024 * 256 + TPB - 1) / TPB, TPB, 0, stream>>>(fc1w, fc1t, 2 * DD, 1024, 2 * DD);
  cvt_wt_kernel<<<(1024 * 1024 + TPB - 1) / TPB, TPB, 0, stream>>>(fc2w, fc2t, 1024, 1024, 1024);
  cvt_bf16_kernel<<<(NG * GENES_P + TPB - 1) / TPB, TPB, 0, stream>>>(exp_in, expbf, NG, GENES, GENES_P);

  // -------- embedding gather --------
  embed_kernel<<<N_NODES * 32 / TPB, TPB, 0, stream>>>(x, emb, bufA);

  // -------- 5 GIN layers --------
  const int cvtN = (int)(ND / TPB);                 // 200000 blocks
  for (int L = 0; L < 5; ++L) {
    hipMemcpyAsync(bufB, bufA, ND * sizeof(float), hipMemcpyDeviceToDevice, stream);
    scatter_kernel<<<N_EDGES * 32 / TPB, TPB, 0, stream>>>(ei, bufA, bufB);
    cvt_bf16_kernel<<<cvtN, TPB, 0, stream>>>(bufB, bufBF, N_NODES, DD, DD);
    run_gemm(bufBF, w1t[L], b1[L], bufA, N_NODES, DD, DD, DD, true, false, stream);
    cvt_bf16_kernel<<<cvtN, TPB, 0, stream>>>(bufA, bufBF, N_NODES, DD, DD);
    run_gemm(bufBF, w2t[L], b2[L], bufB, N_NODES, DD, DD, DD, true, false, stream);
    zero_kernel<<<1, TPB, 0, stream>>>(stats, 256);
    bn_stats_kernel<<<(N_NODES + 1023) / 1024, 128, 0, stream>>>(bufB, stats, N_NODES, 1024);
    bn_apply_kernel<<<cvtN, TPB, 0, stream>>>(bufB, stats, bng[L], bnb[L], bufA, N_NODES);
  }

  // -------- pooling --------
  zero_kernel<<<(NG * DD + TPB - 1) / TPB, TPB, 0, stream>>>(pooled, NG * DD);
  pool_kernel<<<N_NODES * 32 / TPB, TPB, 0, stream>>>(batch, bufA, pooled);

  // -------- head: xd | xt -> fc1 -> fc2 -> fc3 --------
  unsigned short* pooledbf = bufBF;                                   // [G][128]
  unsigned short* gelubf   = bufBF + (size_t)NG * DD;                 // [G][128]
  unsigned short* combbf   = bufBF + (size_t)2 * NG * DD;             // [G][256]
  unsigned short* fc1bf    = bufBF + (size_t)4 * NG * DD;             // [G][1024]

  cvt_bf16_kernel<<<NG * DD / TPB, TPB, 0, stream>>>(pooled, pooledbf, NG, DD, DD);
  run_gemm(pooledbf, fxdt, fxd_b, combbf, NG, DD, DD, 2 * DD, true, true, stream);   // xd -> comb[:,0:128]

  run_gemm(expbf, cw1t, cb1, bufA, NG, DD, GENES_P, DD, false, false, stream);       // exp @ cw1 + cb1
  gelu_bf16_kernel<<<NG * DD / TPB, TPB, 0, stream>>>(bufA, gelubf, NG * DD);
  run_gemm(gelubf, cw2t, cb2, combbf + DD, NG, DD, DD, 2 * DD, false, true, stream); // xt -> comb[:,128:256]

  run_gemm(combbf, fc1t, fc1b, fc1bf, NG, 1024, 2 * DD, 1024, true, true, stream);   // fc1 + relu (bf16)
  run_gemm(fc1bf, fc2t, fc2b, bufB, NG, 1024, 1024, 1024, true, false, stream);      // fc2 + relu (f32)
  fc3_kernel<<<NG / 8, TPB, 0, stream>>>(bufB, fc3w, fc3b, (float*)d_out);
}
